// LSTM_19370302505673
// MI455X (gfx1250) — matrix operand order repaired and checked by reading, not compiled, not run on hardware
//
#include <hip/hip_runtime.h>

// ---------------------------------------------------------------------------
// 2-layer LSTM for MI455X (gfx1250, wave32, WMMA). B=128, C=H=64, S=4096.
//
// Path A (ws >= 320MB):
//   1) convert weights to f16; transpose x -> xT [S][B*C] f16
//   2) k_xgemm: parallel WMMA GEMM xg0 = x@Wih0^T + b0 for ALL timesteps,
//      stored f16 pre-swizzled in the persistent kernel's D-frag layout
//   3) k_lstm_pre: persistent recurrence, 3 GEMMs/step (L0 recurrent,
//      L1 feed-forward, L1 recurrent); xg0 arrives as 4 coalesced
//      global_load_b128 per wave per step.
// Path B (fallback): round-1 fully fused persistent kernel (4 GEMMs/step).
// ---------------------------------------------------------------------------

#define S_LEN  4096
#define BATCH  128
#define HID    64
#define PAD    72              // padded LDS row stride in halfs (144B, 16B-aligned)
#define NW_H   (256 * PAD)     // halfs per weight matrix in LDS
#define NX_H   (BATCH * PAD)   // halfs per x/h tile in LDS

typedef _Float16 half_t;
typedef __attribute__((ext_vector_type(8)))  _Float16 v8h;
typedef __attribute__((ext_vector_type(16))) _Float16 v16h;
typedef __attribute__((ext_vector_type(8)))  float    v8f;

static constexpr int SMEM_B = (4 * NW_H + 2 * NX_H + 2 * NX_H) * 2; // path B LDS
static constexpr int SMEM_A = (3 * NW_H + 2 * NX_H) * 2;            // path A LDS

// ---------------------------------------------------------------------------
// CDNA5 hardware tanh (transcendental unit); sigmoid via tanh identity.
// ---------------------------------------------------------------------------
__device__ __forceinline__ float htanh(float x) {
#if __has_builtin(__builtin_amdgcn_tanhf)
  return __builtin_amdgcn_tanhf(x);
#else
  float y;
  asm volatile("v_tanh_f32 %0, %1" : "=v"(y) : "v"(x));
  return y;
#endif
}
__device__ __forceinline__ float hsig(float x) {
  return 0.5f + 0.5f * htanh(0.5f * x);
}

__device__ __forceinline__ v8f splat8(float v) {
  v8f z;
#pragma unroll
  for (int r = 0; r < 8; ++r) z[r] = v;
  return z;
}

// ---------------------------------------------------------------------------
// WMMA fragment loaders (16x16x32 f16, wave32 layouts per CDNA5 ISA 7.12.2).
// ---------------------------------------------------------------------------
__device__ __forceinline__ v16h load_afrag(const half_t* base, int stride,
                                           int m0, int kt, int lane) {
  const int lm = lane & 15;
  const int lh = lane >> 4;
  const half_t* p = base + (m0 + lm) * stride + kt * 32 + lh * 8;
  v8h lo = *(const v8h*)(p);
  v8h hi = *(const v8h*)(p + 16);
  v16h a;
#pragma unroll
  for (int i = 0; i < 8; ++i) { a[i] = lo[i]; a[i + 8] = hi[i]; }
  return a;
}

__device__ __forceinline__ v16h load_bfrag(const half_t* w, int n0, int kt, int lane) {
  const int lm = lane & 15;
  const int lh = lane >> 4;
  const half_t* p = w + (n0 + lm) * PAD + kt * 32 + lh * 8;
  v8h lo = *(const v8h*)(p);
  v8h hi = *(const v8h*)(p + 16);
  v16h b;
#pragma unroll
  for (int i = 0; i < 8; ++i) { b[i] = lo[i]; b[i + 8] = hi[i]; }
  return b;
}

// Accumulate A[m0.., K=64] x W^T into the wave's 4 gate tiles (n = q*64 + hc).
__device__ __forceinline__ void gemm4(v8f (&acc)[4], const half_t* a_src, int a_stride,
                                      const half_t* w, int m0, int hc, int lane) {
#pragma unroll
  for (int kt = 0; kt < 2; ++kt) {
    const v16h a = load_afrag(a_src, a_stride, m0, kt, lane);
#pragma unroll
    for (int q = 0; q < 4; ++q) {
      const v16h b = load_bfrag(w, q * 64 + hc, kt, lane);
      acc[q] = __builtin_amdgcn_wmma_f32_16x16x32_f16(
          false, a, false, b, (short)0, acc[q], false, false);
    }
  }
}

// Cooperative copy of one timestep tile xT[t] (128x64 f16) into padded LDS.
__device__ __forceinline__ void load_x_tile(half_t* dst, const half_t* __restrict__ xT,
                                            int t, int tid) {
  const int b = tid >> 3, c8 = tid & 7;
  const half_t* src = xT + (size_t)t * (BATCH * HID) + b * HID + c8 * 8;
  *(v8h*)(dst + b * PAD + c8 * 8) = *(const v8h*)src;
  __builtin_prefetch(src + 2 * (BATCH * HID), 0, 1);
}

// ---------------------------------------------------------------------------
// Prep kernel 1: pack all 4 weight matrices ([256,64] f32) into f16.
// ---------------------------------------------------------------------------
__global__ __launch_bounds__(256) void k_convert_w(
    const float* __restrict__ w0, const float* __restrict__ w1,
    const float* __restrict__ w2, const float* __restrict__ w3,
    half_t* __restrict__ wpack) {
  const int idx = blockIdx.x * 256 + threadIdx.x;  // 0..65535
  const float* src = (idx < 16384) ? w0 : (idx < 32768) ? w1
                   : (idx < 49152) ? w2 : w3;
  wpack[idx] = (half_t)src[idx & 16383];
}

// ---------------------------------------------------------------------------
// Prep kernel 2: transpose x [B*C][S] f32  ->  xT [S][B*C] f16 via LDS tiles.
// ---------------------------------------------------------------------------
__global__ __launch_bounds__(256) void k_transpose_x(
    const float* __restrict__ x, half_t* __restrict__ xT) {
  __shared__ half_t tile[64][PAD];
  const int s0  = blockIdx.x * 64;
  const int bc0 = blockIdx.y * 64;
  const int tid = threadIdx.x;
#pragma unroll
  for (int j = 0; j < 4; ++j) {
    const int idx = tid + j * 256;
    const int row = idx >> 4;
    const int c4  = idx & 15;
    float4 v = *(const float4*)(x + (size_t)(bc0 + row) * S_LEN + s0 + c4 * 4);
    tile[c4 * 4 + 0][row] = (half_t)v.x;
    tile[c4 * 4 + 1][row] = (half_t)v.y;
    tile[c4 * 4 + 2][row] = (half_t)v.z;
    tile[c4 * 4 + 3][row] = (half_t)v.w;
  }
  __syncthreads();
#pragma unroll
  for (int j = 0; j < 2; ++j) {
    const int idx  = tid + j * 256;
    const int srow = idx >> 3;
    const int c8   = idx & 7;
    v8h vv = *(const v8h*)(&tile[srow][c8 * 8]);
    *(v8h*)(xT + (size_t)(s0 + srow) * (BATCH * HID) + bc0 + c8 * 8) = vv;
  }
}

// ---------------------------------------------------------------------------
// Path A kernel: xg0[t] = x_t @ Wih0^T + b0 for all t, via WMMA, stored f16
// pre-swizzled as [t][wave][lane][q][r] so the persistent kernel's D-frag
// init is 4 coalesced global_load_b128 per wave.
// Block = 1024 threads (32 waves, same tile map as persistent kernel), 8 t's.
// ---------------------------------------------------------------------------
__global__ __launch_bounds__(1024) void k_xgemm(
    const half_t* __restrict__ xT, const half_t* __restrict__ wpack,
    const float* __restrict__ b0g, half_t* __restrict__ xg0) {
  __shared__ half_t sW[NW_H];
  const int tid = threadIdx.x;
  for (int idx = tid; idx < 256 * 64; idx += 1024)
    sW[(idx >> 6) * PAD + (idx & 63)] = wpack[idx];  // mat 0 = w_ih0

  const int lane = tid & 31, wave = tid >> 5;
  const int m0 = (wave & 7) << 4;
  const int hc = (wave >> 3) << 4;
  const int n  = hc + (lane & 15);
  float bias0[4];
#pragma unroll
  for (int q = 0; q < 4; ++q) bias0[q] = b0g[q * 64 + n];
  __syncthreads();

#pragma unroll 1
  for (int tt = 0; tt < 8; ++tt) {
    const int t = blockIdx.x * 8 + tt;
    v8f acc[4];
#pragma unroll
    for (int q = 0; q < 4; ++q) acc[q] = splat8(bias0[q]);
#pragma unroll
    for (int kt = 0; kt < 2; ++kt) {
      const v16h a = load_afrag(xT + (size_t)t * (BATCH * HID), HID, m0, kt, lane);
#pragma unroll
      for (int q = 0; q < 4; ++q) {
        const v16h b = load_bfrag(sW, q * 64 + hc, kt, lane);
        acc[q] = __builtin_amdgcn_wmma_f32_16x16x32_f16(
            false, a, false, b, (short)0, acc[q], false, false);
      }
    }
    half_t* dst = xg0 + (size_t)t * 32768 + wave * 1024 + lane * 32;
#pragma unroll
    for (int q = 0; q < 4; ++q) {
      v8h o;
#pragma unroll
      for (int r = 0; r < 8; ++r) o[r] = (half_t)acc[q][r];
      *(v8h*)(dst + q * 8) = o;
    }
  }
}

// ---------------------------------------------------------------------------
// Path A persistent recurrence: 3 GEMMs/step; xg0 D-init streamed from global.
// LDS: Whh0, Wih1, Whh1 + h0 + h1.
// ---------------------------------------------------------------------------
__global__ __launch_bounds__(1024) void k_lstm_pre(
    const half_t* __restrict__ xg0, const half_t* __restrict__ wpack,
    const float* __restrict__ b1g, float* __restrict__ out) {
  extern __shared__ char smem_raw[];
  half_t* sW  = (half_t*)smem_raw;        // [3][256][PAD]: whh0, wih1, whh1
  half_t* sH0 = sW + 3 * NW_H;
  half_t* sH1 = sH0 + NX_H;
  const int tid = threadIdx.x;

  for (int idx = tid; idx < 3 * 256 * 64; idx += 1024) {
    const int mat = idx >> 14, r = (idx >> 6) & 255, c = idx & 63;
    sW[mat * NW_H + r * PAD + c] = wpack[16384 + idx];  // mats 1..3
  }
  for (int idx = tid; idx < 2 * NX_H; idx += 1024) sH0[idx] = (half_t)0.f;

  const int lane = tid & 31, wave = tid >> 5;
  const int m0 = (wave & 7) << 4;
  const int hc = (wave >> 3) << 4;
  const int lm = lane & 15, lh = lane >> 4;
  const int n = hc + lm;
  const int mbase = m0 + lh * 8;

  float bias1[4];
#pragma unroll
  for (int q = 0; q < 4; ++q) bias1[q] = b1g[q * 64 + n];

  v8f c0 = {}, c1 = {};
  v8f os0 = {}, os1 = {}, os2 = {};
  const half_t* xgbase = xg0 + wave * 1024 + lane * 32;

  __syncthreads();

  for (int t = 0; t < S_LEN; ++t) {
    // D-init fragments for layer 0 (precomputed x@Wih0^T + b0), coalesced.
    v8h xg[4];
#pragma unroll
    for (int q = 0; q < 4; ++q)
      xg[q] = *(const v8h*)(xgbase + (size_t)t * 32768 + q * 8);
    __builtin_prefetch(xgbase + (size_t)(t + 1) * 32768, 0, 1);

    // ---- layer 0: acc = h0 @ Whh0^T, then += xg ----
    v8f acc[4];
#pragma unroll
    for (int q = 0; q < 4; ++q) acc[q] = splat8(0.f);
    gemm4(acc, sH0, PAD, sW + 0 * NW_H, m0, hc, lane);
    __syncthreads();  // all reads of old sH0 complete
#pragma unroll
    for (int q = 0; q < 4; ++q)
#pragma unroll
      for (int r = 0; r < 8; ++r) acc[q][r] += (float)xg[q][r];
#pragma unroll
    for (int r = 0; r < 8; ++r) {
      const float ig = hsig(acc[0][r]);
      const float fg = hsig(acc[1][r]);
      const float gg = htanh(acc[2][r]);
      const float og = hsig(acc[3][r]);
      const float cn = fg * c0[r] + ig * gg;
      c0[r] = cn;
      sH0[(mbase + r) * PAD + n] = (half_t)(og * htanh(cn));
    }
    __syncthreads();  // new sH0 visible

    // ---- layer 1 ----
#pragma unroll
    for (int q = 0; q < 4; ++q) acc[q] = splat8(bias1[q]);
    gemm4(acc, sH0, PAD, sW + 1 * NW_H, m0, hc, lane);
    gemm4(acc, sH1, PAD, sW + 2 * NW_H, m0, hc, lane);
    __syncthreads();  // all reads of old sH1 complete
    v8f h1n;
#pragma unroll
    for (int r = 0; r < 8; ++r) {
      const float ig = hsig(acc[0][r]);
      const float fg = hsig(acc[1][r]);
      const float gg = htanh(acc[2][r]);
      const float og = hsig(acc[3][r]);
      const float cn = fg * c1[r] + ig * gg;
      c1[r] = cn;
      h1n[r] = og * htanh(cn);
      sH1[(mbase + r) * PAD + n] = (half_t)h1n[r];
    }

    const int tsub = t & 3;
    if (tsub == 0)      os0 = h1n;
    else if (tsub == 1) os1 = h1n;
    else if (tsub == 2) os2 = h1n;
    else {
#pragma unroll
      for (int r = 0; r < 8; ++r) {
        float4 v = make_float4(os0[r], os1[r], os2[r], h1n[r]);
        *(float4*)(out + ((size_t)(mbase + r) * HID + n) * S_LEN + (t - 3)) = v;
      }
    }
  }
}

// ---------------------------------------------------------------------------
// Path B persistent kernel (fully fused, round-1 design).
// ---------------------------------------------------------------------------
__global__ __launch_bounds__(1024) void k_lstm(
    const half_t* __restrict__ xT, const half_t* __restrict__ wpack,
    const float* __restrict__ b0g, const float* __restrict__ b1g,
    float* __restrict__ out) {
  extern __shared__ char smem_raw[];
  half_t* sW  = (half_t*)smem_raw;        // [4][256][PAD]
  half_t* sX  = sW + 4 * NW_H;            // [2][128][PAD]
  half_t* sH0 = sX + 2 * NX_H;
  half_t* sH1 = sH0 + NX_H;
  const int tid = threadIdx.x;

  for (int idx = tid; idx < 4 * 256 * 64; idx += 1024) {
    const int mat = idx >> 14, r = (idx >> 6) & 255, c = idx & 63;
    sW[mat * NW_H + r * PAD + c] = wpack[idx];
  }
  for (int idx = tid; idx < 2 * NX_H; idx += 1024) sH0[idx] = (half_t)0.f;
  load_x_tile(sX, xT, 0, tid);

  const int lane = tid & 31, wave = tid >> 5;
  const int m0 = (wave & 7) << 4;
  const int hc = (wave >> 3) << 4;
  const int lm = lane & 15, lh = lane >> 4;
  const int n = hc + lm;
  const int mbase = m0 + lh * 8;

  float bias0[4], bias1[4];
#pragma unroll
  for (int q = 0; q < 4; ++q) {
    bias0[q] = b0g[q * 64 + n];
    bias1[q] = b1g[q * 64 + n];
  }

  v8f c0 = {}, c1 = {};
  v8f os0 = {}, os1 = {}, os2 = {};

  __syncthreads();

  for (int t = 0; t < S_LEN; ++t) {
    half_t* xcur = sX + (t & 1) * NX_H;
    if (t + 1 < S_LEN) load_x_tile(sX + ((t + 1) & 1) * NX_H, xT, t + 1, tid);

    v8f acc[4];
#pragma unroll
    for (int q = 0; q < 4; ++q) acc[q] = splat8(bias0[q]);
    gemm4(acc, xcur, PAD, sW + 0 * NW_H, m0, hc, lane);
    gemm4(acc, sH0,  PAD, sW + 1 * NW_H, m0, hc, lane);
    __syncthreads();
#pragma unroll
    for (int r = 0; r < 8; ++r) {
      const float ig = hsig(acc[0][r]);
      const float fg = hsig(acc[1][r]);
      const float gg = htanh(acc[2][r]);
      const float og = hsig(acc[3][r]);
      const float cn = fg * c0[r] + ig * gg;
      c0[r] = cn;
      sH0[(mbase + r) * PAD + n] = (half_t)(og * htanh(cn));
    }
    __syncthreads();

#pragma unroll
    for (int q = 0; q < 4; ++q) acc[q] = splat8(bias1[q]);
    gemm4(acc, sH0, PAD, sW + 2 * NW_H, m0, hc, lane);
    gemm4(acc, sH1, PAD, sW + 3 * NW_H, m0, hc, lane);
    __syncthreads();
    v8f h1n;
#pragma unroll
    for (int r = 0; r < 8; ++r) {
      const float ig = hsig(acc[0][r]);
      const float fg = hsig(acc[1][r]);
      const float gg = htanh(acc[2][r]);
      const float og = hsig(acc[3][r]);
      const float cn = fg * c1[r] + ig * gg;
      c1[r] = cn;
      h1n[r] = og * htanh(cn);
      sH1[(mbase + r) * PAD + n] = (half_t)h1n[r];
    }

    const int tsub = t & 3;
    if (tsub == 0)      os0 = h1n;
    else if (tsub == 1) os1 = h1n;
    else if (tsub == 2) os2 = h1n;
    else {
#pragma unroll
      for (int r = 0; r < 8; ++r) {
        float4 v = make_float4(os0[r], os1[r], os2[r], h1n[r]);
        *(float4*)(out + ((size_t)(mbase + r) * HID + n) * S_LEN + (t - 3)) = v;
      }
    }
  }
}

// ---------------------------------------------------------------------------
extern "C" void kernel_launch(void* const* d_in, const int* in_sizes, int n_in,
                              void* d_out, int out_size, void* d_ws, size_t ws_size,
                              hipStream_t stream) {
  (void)in_sizes; (void)n_in; (void)out_size;
  const float* x     = (const float*)d_in[0];
  const float* w_ih0 = (const float*)d_in[1];
  const float* w_hh0 = (const float*)d_in[2];
  const float* b0    = (const float*)d_in[3];
  const float* w_ih1 = (const float*)d_in[4];
  const float* w_hh1 = (const float*)d_in[5];
  const float* b1    = (const float*)d_in[6];
  float* out = (float*)d_out;

  const size_t wpack_bytes = 131072;                      // 4*16384 f16
  const size_t xT_bytes    = (size_t)S_LEN * 8192 * 2;    // 64MB
  const size_t xg0_bytes   = (size_t)S_LEN * 32768 * 2;   // 256MB
  half_t* wpack = (half_t*)d_ws;
  half_t* xT    = (half_t*)((char*)d_ws + wpack_bytes);
  half_t* xg0   = (half_t*)((char*)d_ws + wpack_bytes + xT_bytes);

  k_convert_w<<<256, 256, 0, stream>>>(w_ih0, w_hh0, w_ih1, w_hh1, wpack);
  dim3 tgrid(S_LEN / 64, (BATCH * HID) / 64);
  k_transpose_x<<<tgrid, 256, 0, stream>>>(x, xT);

  if (ws_size >= wpack_bytes + xT_bytes + xg0_bytes) {
    // Path A: precompute layer-0 input gates chip-wide, lighter serial loop.
    k_xgemm<<<S_LEN / 8, 1024, 0, stream>>>(xT, wpack, b0, xg0);
    (void)hipFuncSetAttribute(reinterpret_cast<const void*>(k_lstm_pre),
                              hipFuncAttributeMaxDynamicSharedMemorySize, SMEM_A);
    k_lstm_pre<<<1, 1024, SMEM_A, stream>>>(xg0, wpack, b1, out);
  } else {
    // Path B: fully fused persistent kernel.
    (void)hipFuncSetAttribute(reinterpret_cast<const void*>(k_lstm),
                              hipFuncAttributeMaxDynamicSharedMemorySize, SMEM_B);
    k_lstm<<<1, 1024, SMEM_B, stream>>>(xT, wpack, b0, b1, out);
  }
}
